// PCT_Transformer_17549236371826
// MI455X (gfx1250) — matrix operand order repaired and checked by reading, not compiled, not run on hardware
//
#include <hip/hip_runtime.h>
#include <hip/hip_bf16.h>
#include <math.h>

typedef __attribute__((ext_vector_type(16))) __bf16 v16bf;
typedef __attribute__((ext_vector_type(8)))  __bf16 v8bf;
typedef __attribute__((ext_vector_type(8)))  float  v8f;

#define TOKENS   23040   // 15 * 128 * 12
#define NLIDX    15
#define BSZ      128
#define SEQL     12
#define EDIM     768
#define DHALF    384
#define NHEAD    12
#define HDIM     64
#define NLAY     6
#define E4       3072
#define QKVN     2304
#define BPAD     72      // padded LDS k-stride (elements); 144B rows stay 16B-aligned

__device__ __forceinline__ unsigned short f2bf(float f) {
  union { float f; unsigned int u; } v; v.f = f;
  unsigned int u = v.u;
  u += 0x7FFFu + ((u >> 16) & 1u);
  return (unsigned short)(u >> 16);
}
__device__ __forceinline__ float bf2f(unsigned short h) {
  union { unsigned int u; float f; } v; v.u = ((unsigned int)h) << 16;
  return v.f;
}

// ---------------- weight transpose fp32[K x N] -> bf16[N x K] ----------------
__global__ __launch_bounds__(256) void transpose_w(const float* __restrict__ src,
                                                   unsigned short* __restrict__ dst,
                                                   int K, int N) {
  int total = K * N;
  for (int idx = blockIdx.x * 256 + threadIdx.x; idx < total; idx += gridDim.x * 256) {
    int k = idx / N, n = idx % N;
    dst[(size_t)n * K + k] = f2bf(src[idx]);
  }
}

__global__ __launch_bounds__(256) void pack_qkv_bias(const float* __restrict__ bq,
                                                     const float* __restrict__ bk,
                                                     const float* __restrict__ bv,
                                                     float* __restrict__ out) {
  int idx = blockIdx.x * 256 + threadIdx.x;
  if (idx >= NLAY * QKVN) return;
  int l = idx / QKVN, n = idx % QKVN;
  float v;
  if (n < EDIM)            v = bq[l * EDIM + n];
  else if (n < 2 * EDIM)   v = bk[l * EDIM + n - EDIM];
  else                     v = bv[l * EDIM + n - 2 * EDIM];
  out[idx] = v;
}

// ---------------- build input activations x fp32 [TOKENS x 768] ----------------
__global__ __launch_bounds__(256) void build_x(const float* __restrict__ X,
                                               const float* __restrict__ targets,
                                               const float* __restrict__ omegas,
                                               const float* __restrict__ phases,
                                               const int* __restrict__ perms,
                                               float* __restrict__ x) {
  long long idx = (long long)blockIdx.x * 256 + threadIdx.x;
  if (idx >= (long long)TOKENS * EDIM) return;
  int tok = (int)(idx / EDIM);
  int e   = (int)(idx % EDIM);
  int lidx_i = tok / (BSZ * SEQL);          // 0..14
  int b      = (tok / SEQL) % BSZ;
  int s      = tok % SEQL;
  int lidx   = lidx_i + 1;
  int pb     = perms[lidx_i * BSZ + b];
  float val;
  if (s < 3) {
    int c = s;
    int t = (e < DHALF) ? (16 - lidx) : (15 - lidx);
    int d = (e < DHALF) ? e : (e - DHALF);
    float xv = X[(pb * 16 + t) * 3 + c];
    val = cosf(xv * omegas[c * DHALF + d] + phases[c * DHALF + d]);
  } else if (s < 6) {
    int c = s - 3;
    int t = (17 - lidx) & 15;
    val = targets[(pb * 16 + t) * 3 + c];
  } else {
    val = (float)(16 - lidx);
  }
  x[idx] = val;
}

// ---------------- LayerNorm: fp32 rows -> bf16 rows ----------------
__global__ __launch_bounds__(256) void ln_kernel(const float* __restrict__ x,
                                                 const float* __restrict__ g,
                                                 const float* __restrict__ b,
                                                 unsigned short* __restrict__ h,
                                                 int rows) {
  int wave = threadIdx.x >> 5, lane = threadIdx.x & 31;
  int row = blockIdx.x * 8 + wave;
  if (row >= rows) return;
  const float* xr = x + (size_t)row * EDIM;
  float vals[24];
  float s = 0.f, sq = 0.f;
#pragma unroll
  for (int i = 0; i < 24; i++) {
    float v = xr[lane + 32 * i];
    vals[i] = v; s += v; sq += v * v;
  }
#pragma unroll
  for (int m = 16; m >= 1; m >>= 1) { s += __shfl_xor(s, m, 32); sq += __shfl_xor(sq, m, 32); }
  float mu  = s  * (1.f / EDIM);
  float var = sq * (1.f / EDIM) - mu * mu;
  float inv = rsqrtf(var + 0.001f);
  unsigned short* hr = h + (size_t)row * EDIM;
#pragma unroll
  for (int i = 0; i < 24; i++) {
    int e = lane + 32 * i;
    hr[e] = f2bf((vals[i] - mu) * inv * g[e] + b[e]);
  }
}

// ---------------- bf16 GEMM via v_wmma_f32_16x16x32_bf16 + async B staging ----------------
// A: [M x K] bf16 row-major.  Bt: [N x K] bf16 (B transposed).  bias: [N] fp32 or null.
// mode 0: store bf16; mode 1: relu + store bf16; mode 2: outF += C (residual add)
__device__ __forceinline__ v16bf ld_a_frag(const unsigned short* p) {
  v8bf lo = *reinterpret_cast<const v8bf*>(p);        // K .. K+7
  v8bf hi = *reinterpret_cast<const v8bf*>(p + 16);   // K+16 .. K+23
  v16bf r;
#pragma unroll
  for (int i = 0; i < 8; i++) { r[i] = lo[i]; r[i + 8] = hi[i]; }
  return r;
}
__device__ __forceinline__ v16bf ld_b_frag(const unsigned short* p) {
  v8bf lo = *reinterpret_cast<const v8bf*>(p);        // K .. K+7
  v8bf hi = *reinterpret_cast<const v8bf*>(p + 16);   // K+16 .. K+23
  v16bf r;
#pragma unroll
  for (int i = 0; i < 8; i++) { r[i] = lo[i]; r[i + 8] = hi[i]; }
  return r;
}

__global__ __launch_bounds__(256) void gemm_bf16_wmma(const unsigned short* __restrict__ A,
                                                      const unsigned short* __restrict__ Bt,
                                                      const float* __restrict__ bias,
                                                      unsigned short* __restrict__ outB,
                                                      float* __restrict__ outF,
                                                      int M, int N, int K, int mode) {
  // Double-buffered B tile: [buf][n(64)][k(64 used, BPAD stride)]
  __shared__ __align__(16) unsigned short Bsh[2][64][BPAD];

  const int tid     = threadIdx.x;
  const int lane    = tid & 31;
  const int wave    = tid >> 5;
  const int m0      = blockIdx.y * 128 + wave * 16;
  const int n0      = blockIdx.x * 64;
  const int halfsel = lane >> 4;       // 0: lanes 0-15, 1: lanes 16-31
  const int lm      = lane & 15;

  const unsigned short* arow = A + (size_t)(m0 + lm) * K + halfsel * 8;
  const int bkb = halfsel * 8;         // element offset inside a 32-k step

  const unsigned lds_base = (unsigned)(size_t)&Bsh[0][0][0];

  // Issue one 64-wide K chunk of the B tile (64 rows x 128B) into LDS buffer `buf`
  // via async global->LDS copies (ASYNCcnt tracked). 512 x 16B transfers, 2/thread.
  auto issue_chunk = [&](int k0c, int buf) {
#pragma unroll
    for (int rep = 0; rep < 2; rep++) {
      int tr   = tid + rep * 256;          // 0..511
      int rn   = tr >> 3;                  // row 0..63
      int part = tr & 7;                   // 16B piece within the 128B row
      const unsigned short* gp = Bt + (size_t)(n0 + rn) * K + k0c + part * 8;
      unsigned lds = lds_base + (unsigned)(buf * (64 * BPAD * 2) + rn * (BPAD * 2) + part * 16);
      asm volatile("global_load_async_to_lds_b128 %0, %1, off"
                   :: "v"(lds), "v"(gp)
                   : "memory");
    }
  };

  v8f acc[4];
#pragma unroll
  for (int t = 0; t < 4; t++) acc[t] = (v8f){};

  issue_chunk(0, 0);
  asm volatile("s_wait_asynccnt 0x0" ::: "memory");
  __syncthreads();

  for (int k0 = 0; k0 < K; k0 += 64) {
    int cur = (k0 >> 6) & 1;
    if (k0 + 64 < K) issue_chunk(k0 + 64, 1 - cur);   // overlap fill of next chunk
#pragma unroll
    for (int ks = 0; ks < 64; ks += 32) {
      // Batch all fragment loads first so the WMMAs can issue back-to-back
      // behind a single dependency wait.
      v16bf av  = ld_a_frag(arow + k0 + ks);
      v16bf bv0 = ld_b_frag(&Bsh[cur][ 0 + lm][ks + bkb]);
      v16bf bv1 = ld_b_frag(&Bsh[cur][16 + lm][ks + bkb]);
      v16bf bv2 = ld_b_frag(&Bsh[cur][32 + lm][ks + bkb]);
      v16bf bv3 = ld_b_frag(&Bsh[cur][48 + lm][ks + bkb]);
      acc[0] = __builtin_amdgcn_wmma_f32_16x16x32_bf16(false, av, false, bv0, (short)0, acc[0], false, false);
      acc[1] = __builtin_amdgcn_wmma_f32_16x16x32_bf16(false, av, false, bv1, (short)0, acc[1], false, false);
      acc[2] = __builtin_amdgcn_wmma_f32_16x16x32_bf16(false, av, false, bv2, (short)0, acc[2], false, false);
      acc[3] = __builtin_amdgcn_wmma_f32_16x16x32_bf16(false, av, false, bv3, (short)0, acc[3], false, false);
    }
    asm volatile("s_wait_asynccnt 0x0" ::: "memory");
    __syncthreads();
  }

#pragma unroll
  for (int t = 0; t < 4; t++) {
    int n = n0 + t * 16 + lm;
    float bsv = bias ? bias[n] : 0.f;
#pragma unroll
    for (int r = 0; r < 8; r++) {
      int m = m0 + halfsel * 8 + r;
      float val = acc[t][r] + bsv;
      size_t o = (size_t)m * N + n;
      if (mode == 2) {
        outF[o] += val;
      } else {
        if (mode == 1 && val < 0.f) val = 0.f;
        outB[o] = f2bf(val);
      }
    }
  }
}

// ---------------- attention: one block per (lidx, b, head) ----------------
__global__ __launch_bounds__(128) void attn_kernel(const unsigned short* __restrict__ qkv,
                                                   unsigned short* __restrict__ obuf) {
  __shared__ float q[SEQL][HDIM], k[SEQL][HDIM], v[SEQL][HDIM];
  __shared__ float att[SEQL][SEQL];
  int bid  = blockIdx.x;
  int hh   = bid % NHEAD;
  int tok0 = (bid / NHEAD) * SEQL;
  int tid  = threadIdx.x;

  for (int t = tid; t < SEQL * HDIM; t += 128) {
    int s = t >> 6, d = t & 63;
    size_t base = (size_t)(tok0 + s) * QKVN + hh * HDIM + d;
    q[s][d] = bf2f(qkv[base]);
    k[s][d] = bf2f(qkv[base + EDIM]);
    v[s][d] = bf2f(qkv[base + 2 * EDIM]);
  }
  __syncthreads();
  if (tid < SEQL * SEQL) {
    int i = tid / SEQL, j = tid % SEQL;
    float s = 0.f;
#pragma unroll
    for (int d = 0; d < HDIM; d++) s += q[i][d] * k[j][d];
    att[i][j] = s * 0.125f;   // 1/sqrt(64)
  }
  __syncthreads();
  if (tid < SEQL) {
    float mx = -1e30f;
#pragma unroll
    for (int j = 0; j < SEQL; j++) mx = fmaxf(mx, att[tid][j]);
    float sum = 0.f;
#pragma unroll
    for (int j = 0; j < SEQL; j++) { float e = __expf(att[tid][j] - mx); att[tid][j] = e; sum += e; }
    float inv = 1.f / sum;
#pragma unroll
    for (int j = 0; j < SEQL; j++) att[tid][j] *= inv;
  }
  __syncthreads();
  for (int t = tid; t < SEQL * HDIM; t += 128) {
    int i = t >> 6, d = t & 63;
    float s = 0.f;
#pragma unroll
    for (int j = 0; j < SEQL; j++) s += att[i][j] * v[j][d];
    obuf[(size_t)(tok0 + i) * EDIM + hh * HDIM + d] = f2bf(s);
  }
}

// ---------------- output head: logits[lidx,b,3] ----------------
__global__ __launch_bounds__(256) void logits_kernel(const unsigned short* __restrict__ h,
                                                     const float* __restrict__ Wout,
                                                     const float* __restrict__ bout,
                                                     float* __restrict__ logits) {
  __shared__ float r0[256], r1[256], r2[256];
  int bh = blockIdx.x;                      // lidx_i*128 + b
  int tid = threadIdx.x;
  const unsigned short* hrow = h + (size_t)bh * (SEQL * EDIM);
  float p0 = 0.f, p1 = 0.f, p2 = 0.f;
  for (int idx = tid; idx < SEQL * EDIM; idx += 256) {
    float hv = bf2f(hrow[idx]);
    p0 += hv * Wout[idx * 3 + 0];
    p1 += hv * Wout[idx * 3 + 1];
    p2 += hv * Wout[idx * 3 + 2];
  }
  r0[tid] = p0; r1[tid] = p1; r2[tid] = p2;
  __syncthreads();
  for (int s = 128; s > 0; s >>= 1) {
    if (tid < s) { r0[tid] += r0[tid + s]; r1[tid] += r1[tid + s]; r2[tid] += r2[tid + s]; }
    __syncthreads();
  }
  if (tid == 0) {
    logits[bh * 3 + 0] = r0[0] + bout[0];
    logits[bh * 3 + 1] = r1[0] + bout[1];
    logits[bh * 3 + 2] = r2[0] + bout[2];
  }
}

// ---------------- cost matrix ----------------
__global__ __launch_bounds__(256) void cost_kernel(const float* __restrict__ logits,
                                                   const float* __restrict__ targets,
                                                   float* __restrict__ cost) {
  int idx = blockIdx.x * 256 + threadIdx.x;
  if (idx >= NLIDX * BSZ * BSZ) return;
  int lidx_i = idx / (BSZ * BSZ);
  int r = idx % (BSZ * BSZ);
  int i = r >> 7, j = r & 127;
  int lidx = lidx_i + 1;
  float s = 0.f;
#pragma unroll
  for (int c = 0; c < 3; c++) {
    float d = logits[(lidx_i * BSZ + i) * 3 + c] - targets[(j * 16 + (16 - lidx)) * 3 + c];
    s += d * d;
  }
  cost[idx] = sqrtf(s + 1e-12f);
}

// ---------------- greedy match + loss, one block per lidx ----------------
__global__ __launch_bounds__(256) void match_loss_kernel(const float* __restrict__ cost,
                                                         const float* __restrict__ logits,
                                                         const float* __restrict__ targets,
                                                         float* __restrict__ losses) {
  __shared__ int rowTaken[BSZ], colTaken[BSZ], rowMatch[BSZ];
  __shared__ float rv[256];
  __shared__ int   ri[256];
  int lidx_i = blockIdx.x, lidx = lidx_i + 1;
  int tid = threadIdx.x;
  const float* C = cost + (size_t)lidx_i * BSZ * BSZ;
  for (int i = tid; i < BSZ; i += 256) { rowTaken[i] = 0; colTaken[i] = 0; rowMatch[i] = 0; }
  __syncthreads();
  for (int it = 0; it < BSZ; ++it) {
    float bv = INFINITY; int bi = 1 << 30;
    for (int t = tid; t < BSZ * BSZ; t += 256) {
      int i = t >> 7, j = t & 127;
      if (!rowTaken[i] && !colTaken[j]) {
        float c = C[t];
        if (c < bv || (c == bv && t < bi)) { bv = c; bi = t; }
      }
    }
    rv[tid] = bv; ri[tid] = bi;
    __syncthreads();
    for (int s = 128; s > 0; s >>= 1) {
      if (tid < s) {
        float ov = rv[tid + s]; int oi = ri[tid + s];
        if (ov < rv[tid] || (ov == rv[tid] && oi < ri[tid])) { rv[tid] = ov; ri[tid] = oi; }
      }
      __syncthreads();
    }
    if (tid == 0) {
      int t = ri[0]; int i = t >> 7, j = t & 127;
      rowTaken[i] = 1; colTaken[j] = 1; rowMatch[i] = j;
    }
    __syncthreads();
  }
  // loss: mean over i,c of (logits[rowMatch[i]] - tgt[i])^2
  float part = 0.f;
  const float* LG = logits + lidx_i * BSZ * 3;
  for (int i = tid; i < BSZ; i += 256) {
    int mi = rowMatch[i];
#pragma unroll
    for (int c = 0; c < 3; c++) {
      float d = LG[mi * 3 + c] - targets[(i * 16 + (16 - lidx)) * 3 + c];
      part += d * d;
    }
  }
  rv[tid] = part;
  __syncthreads();
  for (int s = 128; s > 0; s >>= 1) {
    if (tid < s) rv[tid] += rv[tid + s];
    __syncthreads();
  }
  if (tid == 0) losses[lidx_i] = rv[0] / (BSZ * 3.0f);
}

__global__ void final_reduce(const float* __restrict__ losses, float* __restrict__ out) {
  if (threadIdx.x == 0 && blockIdx.x == 0) {
    float s = 0.f;
    for (int i = 0; i < NLIDX; i++) s += losses[i];
    out[0] = s / (float)NLIDX;
  }
}

// =========================== host launcher ===========================
extern "C" void kernel_launch(void* const* d_in, const int* in_sizes, int n_in,
                              void* d_out, int out_size, void* d_ws, size_t ws_size,
                              hipStream_t stream) {
  const float* X       = (const float*)d_in[0];
  const float* targets = (const float*)d_in[1];
  const float* omegas  = (const float*)d_in[2];
  const float* phases  = (const float*)d_in[3];
  const int*   perms   = (const int*)  d_in[4];
  const float* Wq   = (const float*)d_in[5];
  const float* bq   = (const float*)d_in[6];
  const float* Wk   = (const float*)d_in[7];
  const float* bk   = (const float*)d_in[8];
  const float* Wv   = (const float*)d_in[9];
  const float* bv   = (const float*)d_in[10];
  const float* Wo   = (const float*)d_in[11];
  const float* bo   = (const float*)d_in[12];
  const float* ln1g = (const float*)d_in[13];
  const float* ln1b = (const float*)d_in[14];
  const float* ln2g = (const float*)d_in[15];
  const float* ln2b = (const float*)d_in[16];
  const float* W1   = (const float*)d_in[17];
  const float* b1f  = (const float*)d_in[18];
  const float* W2   = (const float*)d_in[19];
  const float* b2f  = (const float*)d_in[20];
  const float* lnfg = (const float*)d_in[21];
  const float* lnfb = (const float*)d_in[22];
  const float* Wout = (const float*)d_in[23];
  const float* bout = (const float*)d_in[24];
  float* out = (float*)d_out;

  char* base = (char*)d_ws;
  size_t off = 0;
  auto take = [&](size_t bytes) -> char* {
    char* p = base + off;
    off += (bytes + 255) & ~(size_t)255;
    return p;
  };
  float*          x_act  = (float*)         take((size_t)TOKENS * EDIM * 4);
  unsigned short* hbuf   = (unsigned short*)take((size_t)TOKENS * EDIM * 2);
  unsigned short* ubuf   = (unsigned short*)take((size_t)TOKENS * E4 * 2);    // qkv / mlp-mid union
  unsigned short* obuf   = (unsigned short*)take((size_t)TOKENS * EDIM * 2);
  unsigned short* qkvT   = (unsigned short*)take((size_t)NLAY * QKVN * EDIM * 2);
  float*          qkvB   = (float*)         take((size_t)NLAY * QKVN * 4);
  unsigned short* WoT    = (unsigned short*)take((size_t)NLAY * EDIM * EDIM * 2);
  unsigned short* W1T    = (unsigned short*)take((size_t)NLAY * E4 * EDIM * 2);
  unsigned short* W2T    = (unsigned short*)take((size_t)NLAY * EDIM * E4 * 2);
  float*          logits = (float*)         take((size_t)NLIDX * BSZ * 3 * 4);
  float*          cost   = (float*)         take((size_t)NLIDX * BSZ * BSZ * 4);
  float*          losses = (float*)         take(64);
  (void)ws_size; (void)in_sizes; (void)n_in; (void)out_size;

  // ---- weight prep (transpose to [N x K] bf16, fuse qkv bias) ----
  auto tgrid = [](int elems) { return dim3((elems + 255) / 256); };
  for (int l = 0; l < NLAY; l++) {
    transpose_w<<<tgrid(EDIM * EDIM), 256, 0, stream>>>(Wq + (size_t)l * EDIM * EDIM, qkvT + (size_t)l * QKVN * EDIM,                           EDIM, EDIM);
    transpose_w<<<tgrid(EDIM * EDIM), 256, 0, stream>>>(Wk + (size_t)l * EDIM * EDIM, qkvT + (size_t)l * QKVN * EDIM + (size_t)EDIM * EDIM,     EDIM, EDIM);
    transpose_w<<<tgrid(EDIM * EDIM), 256, 0, stream>>>(Wv + (size_t)l * EDIM * EDIM, qkvT + (size_t)l * QKVN * EDIM + (size_t)2 * EDIM * EDIM, EDIM, EDIM);
    transpose_w<<<tgrid(EDIM * EDIM), 256, 0, stream>>>(Wo + (size_t)l * EDIM * EDIM, WoT + (size_t)l * EDIM * EDIM, EDIM, EDIM);
    transpose_w<<<tgrid(EDIM * E4),   256, 0, stream>>>(W1 + (size_t)l * EDIM * E4,   W1T + (size_t)l * E4 * EDIM,   EDIM, E4);
    transpose_w<<<tgrid(E4 * EDIM),   256, 0, stream>>>(W2 + (size_t)l * E4 * EDIM,   W2T + (size_t)l * EDIM * E4,   E4,   EDIM);
  }
  pack_qkv_bias<<<tgrid(NLAY * QKVN), 256, 0, stream>>>(bq, bk, bv, qkvB);

  // ---- build inputs for all 15 lidx problems ----
  build_x<<<tgrid(TOKENS * EDIM), 256, 0, stream>>>(X, targets, omegas, phases, perms, x_act);

  const dim3 blk(256);
  const dim3 ln_grid(TOKENS / 8);
  const int MT = TOKENS;

  for (int l = 0; l < NLAY; l++) {
    // attention block
    ln_kernel<<<ln_grid, blk, 0, stream>>>(x_act, ln1g + l * EDIM, ln1b + l * EDIM, hbuf, MT);
    gemm_bf16_wmma<<<dim3(QKVN / 64, MT / 128), blk, 0, stream>>>(
        hbuf, qkvT + (size_t)l * QKVN * EDIM, qkvB + (size_t)l * QKVN,
        ubuf, nullptr, MT, QKVN, EDIM, 0);
    attn_kernel<<<dim3(NLIDX * BSZ * NHEAD), dim3(128), 0, stream>>>(ubuf, obuf);
    gemm_bf16_wmma<<<dim3(EDIM / 64, MT / 128), blk, 0, stream>>>(
        obuf, WoT + (size_t)l * EDIM * EDIM, bo + l * EDIM,
        nullptr, x_act, MT, EDIM, EDIM, 2);
    // MLP block
    ln_kernel<<<ln_grid, blk, 0, stream>>>(x_act, ln2g + l * EDIM, ln2b + l * EDIM, hbuf, MT);
    gemm_bf16_wmma<<<dim3(E4 / 64, MT / 128), blk, 0, stream>>>(
        hbuf, W1T + (size_t)l * E4 * EDIM, b1f + l * E4,
        ubuf, nullptr, MT, E4, EDIM, 1);
    gemm_bf16_wmma<<<dim3(EDIM / 64, MT / 128), blk, 0, stream>>>(
        ubuf, W2T + (size_t)l * EDIM * E4, b2f + l * EDIM,
        nullptr, x_act, MT, EDIM, E4, 2);
  }

  // final LN + head + matching + loss
  ln_kernel<<<ln_grid, blk, 0, stream>>>(x_act, lnfg, lnfb, hbuf, MT);
  logits_kernel<<<dim3(NLIDX * BSZ), blk, 0, stream>>>(hbuf, Wout, bout, logits);
  cost_kernel<<<tgrid(NLIDX * BSZ * BSZ), blk, 0, stream>>>(logits, targets, cost);
  match_loss_kernel<<<dim3(NLIDX), blk, 0, stream>>>(cost, logits, targets, losses);
  final_reduce<<<1, 32, 0, stream>>>(losses, out);
}
